// seq2seq_40647570489545
// MI455X (gfx1250) — matrix operand order repaired and checked by reading, not compiled, not run on hardware
//
#include <hip/hip_runtime.h>
#include <hip/hip_bf16.h>
#include <math.h>

typedef __attribute__((ext_vector_type(16))) _Float16 v16h;
typedef __attribute__((ext_vector_type(8)))  _Float16 v8h;
typedef __attribute__((ext_vector_type(8)))  float    v8f;

// ---------------------------------------------------------------------------
// f32 -> f16 conversion for weights (one-time per launch; weights then live in L2)
// ---------------------------------------------------------------------------
__global__ void cvt_f32_f16(const float* __restrict__ in, _Float16* __restrict__ out, int n) {
    int i = blockIdx.x * blockDim.x + threadIdx.x;
    if (i < n) out[i] = (_Float16)in[i];
}

// ---------------------------------------------------------------------------
// Embedding gather (f32 table -> f16 activation rows, optional column offset
// for writing into the [emb | ctx] concat buffer)
// ---------------------------------------------------------------------------
__global__ void gather_embed_f16(_Float16* __restrict__ out, int out_stride, int col0,
                                 const float* __restrict__ table,
                                 const int* __restrict__ toks, int H, int B) {
    int i = blockIdx.x * blockDim.x + threadIdx.x;
    if (i >= B * H) return;
    int b = i / H, h = i - b * H;
    int tok = toks[b];
    out[(size_t)b * out_stride + col0 + h] = (_Float16)table[(size_t)tok * H + h];
}

// ---------------------------------------------------------------------------
// Decoder input tokens: [ones(1,B); target_seq[:-1]]
// ---------------------------------------------------------------------------
__global__ void prep_dec_tokens(const int* __restrict__ target_seq, int* __restrict__ dec_tok,
                                int T, int B) {
    int i = blockIdx.x * blockDim.x + threadIdx.x;
    if (i >= T * B) return;
    dec_tok[i] = (i < B) ? 1 : target_seq[i - B];
}

// ---------------------------------------------------------------------------
// Zero-init of recurrent state + stat accumulators (ws is poisoned by harness)
// ---------------------------------------------------------------------------
__global__ void init_state(float* __restrict__ h, _Float16* __restrict__ h16,
                           float* __restrict__ lossPerB, int* __restrict__ correctPerB,
                           int* __restrict__ wrong, int BH, int B) {
    int i = blockIdx.x * blockDim.x + threadIdx.x;
    if (i < BH) { h[i] = 0.0f; h16[i] = (_Float16)0.0f; }
    if (i < B)  { lossPerB[i] = 0.0f; correctPerB[i] = 0; wrong[i] = 0; }
}

// ---------------------------------------------------------------------------
// WMMA GEMM:  C[M,N] = A[M,K] (f16, row-major) * W[N,K]^T (f16, row-major) + bias
// One wave computes a 32x32 output patch via 4 accumulators; block = 8 waves
// arranged 2(M) x 4(N) -> 64 x 128 block tile.  K in steps of 32 using
// v_wmma_f32_16x16x32_f16.  act: 0=none, 1=relu, 2=tanh.  Cf (f32) and/or
// Ch (f16) outputs.
// ---------------------------------------------------------------------------
__global__ __launch_bounds__(256)
void wmma_gemm_f16(const _Float16* __restrict__ A, int lda,
                   const _Float16* __restrict__ W, int ldw,
                   const float* __restrict__ bias,
                   float* __restrict__ Cf, _Float16* __restrict__ Ch, int ldc,
                   int K, int act) {
    const int lane  = threadIdx.x & 31;
    const int wave  = threadIdx.x >> 5;
    const int wm    = wave & 1;   // 2 waves along M
    const int wn    = wave >> 1;  // 4 waves along N
    const int m0    = blockIdx.x * 64 + wm * 32;
    const int n0    = blockIdx.y * 128 + wn * 32;
    const int l15   = lane & 15;
    const int lhi   = lane >> 4;

    // A fragment (16x32 f16): lane holds row (m0+l15), 16 contiguous K at 16*lhi
    const _Float16* a0p = A + (size_t)(m0 + l15) * lda + lhi * 16;
    const _Float16* a1p = a0p + (size_t)16 * lda;
    // B fragment (32x16 f16): lane holds col (n0+l15), 16 contiguous K at 16*lhi
    const _Float16* b0p = W + (size_t)(n0 + l15) * ldw + lhi * 16;
    const _Float16* b1p = b0p + (size_t)16 * ldw;

    v8f c00 = {}, c01 = {}, c10 = {}, c11 = {};

    for (int k0 = 0; k0 < K; k0 += 32) {
        v8h a0lo = *(const v8h*)(a0p + k0);
        v8h a0hi = *(const v8h*)(a0p + k0 + 8);
        v8h a1lo = *(const v8h*)(a1p + k0);
        v8h a1hi = *(const v8h*)(a1p + k0 + 8);
        v16h b0  = *(const v16h*)(b0p + k0);
        v16h b1  = *(const v16h*)(b1p + k0);
        v16h a0, a1;
#pragma unroll
        for (int i = 0; i < 8; ++i) {
            a0[i] = a0lo[i]; a0[i + 8] = a0hi[i];
            a1[i] = a1lo[i]; a1[i + 8] = a1hi[i];
        }
        c00 = __builtin_amdgcn_wmma_f32_16x16x32_f16(false, a0, false, b0, (short)0, c00, false, false);
        c01 = __builtin_amdgcn_wmma_f32_16x16x32_f16(false, a0, false, b1, (short)0, c01, false, false);
        c10 = __builtin_amdgcn_wmma_f32_16x16x32_f16(false, a1, false, b0, (short)0, c10, false, false);
        c11 = __builtin_amdgcn_wmma_f32_16x16x32_f16(false, a1, false, b1, (short)0, c11, false, false);
    }

    auto store_tile = [&](int mt, int nt, v8f c) {
        int col = n0 + nt * 16 + l15;
        float bv = bias ? bias[col] : 0.0f;
#pragma unroll
        for (int r = 0; r < 8; ++r) {
            int row = m0 + mt * 16 + r + lhi * 8;
            float v = c[r] + bv;
            if (act == 1) v = fmaxf(v, 0.0f);
            else if (act == 2) v = tanhf(v);
            size_t off = (size_t)row * ldc + col;
            if (Cf) Cf[off] = v;
            if (Ch) Ch[off] = (_Float16)v;
        }
    };
    store_tile(0, 0, c00); store_tile(0, 1, c01);
    store_tile(1, 0, c10); store_tile(1, 1, c11);
}

// ---------------------------------------------------------------------------
// GRU gate fusion: r/z/n sigmoid+tanh, masked update, writes f32 + f16 state
// and (encoder only) the masked enc_out row in f16.
// ---------------------------------------------------------------------------
__global__ void gru_update(const float* __restrict__ gi, const float* __restrict__ gh,
                           float* __restrict__ h, _Float16* __restrict__ h16,
                           _Float16* __restrict__ enc_out_t,
                           const int* __restrict__ len, int t, int B, int H) {
    int i = blockIdx.x * blockDim.x + threadIdx.x;
    if (i >= B * H) return;
    int b = i / H, j = i - b * H;
    size_t g = (size_t)b * 3 * H + j;
    float r = 1.0f / (1.0f + expf(-(gi[g] + gh[g])));
    float z = 1.0f / (1.0f + expf(-(gi[g + H] + gh[g + H])));
    float n = tanhf(gi[g + 2 * H] + r * gh[g + 2 * H]);
    float hold = h[i];
    float hn = (1.0f - z) * n + z * hold;
    bool m = len ? (t < len[b]) : true;
    float hv = m ? hn : hold;
    h[i] = hv;
    h16[i] = (_Float16)hv;
    if (enc_out_t) enc_out_t[i] = m ? (_Float16)hv : (_Float16)0.0f;
}

// ---------------------------------------------------------------------------
// Attention scores: e[b,s] = dot(enc_out[s,b,:], ah[b,:]); one wave per (b,s),
// 16 f16 per lane (32 lanes cover H=512 contiguously), wave32 shuffle reduce.
// ---------------------------------------------------------------------------
__global__ __launch_bounds__(256)
void att_scores(const _Float16* __restrict__ enc, const _Float16* __restrict__ ah,
                float* __restrict__ e, int S, int B, int H) {
    int gw = blockIdx.x * (blockDim.x >> 5) + (threadIdx.x >> 5);
    if (gw >= B * S) return;
    int b = gw / S, s = gw - b * S;
    int lane = threadIdx.x & 31;
    const _Float16* ep = enc + ((size_t)s * B + b) * H;
    const _Float16* ap = ah + (size_t)b * H;
    float acc = 0.0f;
    for (int h0 = lane * 16; h0 < H; h0 += 32 * 16) {
        v16h ev = *(const v16h*)(ep + h0);
        v16h av = *(const v16h*)(ap + h0);
#pragma unroll
        for (int i = 0; i < 16; ++i) acc += (float)ev[i] * (float)av[i];
    }
#pragma unroll
    for (int off = 16; off > 0; off >>= 1) acc += __shfl_xor(acc, off, 32);
    if (lane == 0) e[(size_t)b * S + s] = acc;
}

// ---------------------------------------------------------------------------
// Row softmax over S (<=256), one block per b.
// ---------------------------------------------------------------------------
__global__ __launch_bounds__(256)
void softmax_rows(const float* __restrict__ e, float* __restrict__ att, int S) {
    __shared__ float red[256];
    int b = blockIdx.x, i = threadIdx.x;
    float v = (i < S) ? e[(size_t)b * S + i] : -1e30f;
    red[i] = v; __syncthreads();
    for (int st = 128; st; st >>= 1) { if (i < st) red[i] = fmaxf(red[i], red[i + st]); __syncthreads(); }
    float mx = red[0]; __syncthreads();
    float p = (i < S) ? expf(v - mx) : 0.0f;
    red[i] = p; __syncthreads();
    for (int st = 128; st; st >>= 1) { if (i < st) red[i] += red[i + st]; __syncthreads(); }
    float inv = 1.0f / red[0];
    if (i < S) att[(size_t)b * S + i] = p * inv;
}

// ---------------------------------------------------------------------------
// Context: ctx[b,h] = sum_s att[b,s] * enc_out[s,b,h], written as f16 into the
// second half of the [emb|ctx] concat buffer.
// ---------------------------------------------------------------------------
__global__ __launch_bounds__(256)
void att_context(const _Float16* __restrict__ enc, const float* __restrict__ att,
                 _Float16* __restrict__ xcat, int S, int B, int H) {
    __shared__ float a_sh[256];
    int b = blockIdx.x, tid = threadIdx.x;
    a_sh[tid] = (tid < S) ? att[(size_t)b * S + tid] : 0.0f;
    __syncthreads();
    for (int h = tid; h < H; h += blockDim.x) {
        float acc = 0.0f;
        for (int s = 0; s < S; ++s)
            acc += a_sh[s] * (float)enc[((size_t)s * B + b) * H + h];
        xcat[(size_t)b * 2 * H + H + h] = (_Float16)acc;
    }
}

// ---------------------------------------------------------------------------
// Output softmax statistics: argmax (first-occurrence ties), target log-prob,
// per-b deterministic accumulation (single writer per (t,b), stream-ordered).
// ---------------------------------------------------------------------------
__global__ __launch_bounds__(256)
void out_softmax_stats(const float* __restrict__ logits, const int* __restrict__ tgt_t,
                       float* __restrict__ out_inf_t, float* __restrict__ lossPerB,
                       int* __restrict__ correctPerB, int* __restrict__ wrong, int V) {
    __shared__ float sv[256];
    __shared__ int   si[256];
    __shared__ float ss[256];
    int b = blockIdx.x, tid = threadIdx.x;
    const float* row = logits + (size_t)b * V;
    float best = -1e30f; int bidx = 0x7fffffff;
    for (int v = tid; v < V; v += blockDim.x) {
        float x = row[v];
        if (x > best || (x == best && v < bidx)) { best = x; bidx = v; }
    }
    sv[tid] = best; si[tid] = bidx; __syncthreads();
    for (int st = 128; st; st >>= 1) {
        if (tid < st) {
            float o = sv[tid + st]; int oi = si[tid + st];
            if (o > sv[tid] || (o == sv[tid] && oi < si[tid])) { sv[tid] = o; si[tid] = oi; }
        }
        __syncthreads();
    }
    float mx = sv[0]; int pred = si[0]; __syncthreads();
    float lsum = 0.0f;
    for (int v = tid; v < V; v += blockDim.x) lsum += expf(row[v] - mx);
    ss[tid] = lsum; __syncthreads();
    for (int st = 128; st; st >>= 1) { if (tid < st) ss[tid] += ss[tid + st]; __syncthreads(); }
    if (tid == 0) {
        int tg = tgt_t[b];
        float p = expf(row[tg] - mx) / ss[0];
        p = fmaxf(p, 1e-10f);
        lossPerB[b] += -logf(p);
        out_inf_t[b] = (float)pred;
        if (pred == tg) correctPerB[b] += 1; else wrong[b] = 1;
    }
}

// ---------------------------------------------------------------------------
// Final scalar reduction: loss mean, sequence accuracy, token accuracy.
// ---------------------------------------------------------------------------
__global__ __launch_bounds__(256)
void finalize_kernel(const float* __restrict__ lossPerB, const int* __restrict__ correctPerB,
                     const int* __restrict__ wrong, float* __restrict__ out, int TB, int B) {
    __shared__ float fs[256];
    __shared__ int   cs[256];
    __shared__ int   ws[256];
    int tid = threadIdx.x;
    fs[tid] = (tid < B) ? lossPerB[tid] : 0.0f;
    cs[tid] = (tid < B) ? correctPerB[tid] : 0;
    ws[tid] = (tid < B && wrong[tid] == 0) ? 1 : 0;
    __syncthreads();
    for (int st = 128; st; st >>= 1) {
        if (tid < st) { fs[tid] += fs[tid + st]; cs[tid] += cs[tid + st]; ws[tid] += ws[tid + st]; }
        __syncthreads();
    }
    if (tid == 0) {
        out[TB]     = fs[0] / (float)TB;          // loss
        out[TB + 1] = (float)ws[0] / (float)B;    // acc (all-correct sequences)
        out[TB + 2] = (float)cs[0] / (float)TB;   // all_acc (token accuracy)
    }
}

// ---------------------------------------------------------------------------
// Host orchestration
// ---------------------------------------------------------------------------
extern "C" void kernel_launch(void* const* d_in, const int* in_sizes, int n_in,
                              void* d_out, int out_size, void* d_ws, size_t ws_size,
                              hipStream_t stream) {
    (void)n_in; (void)out_size; (void)ws_size;
    const int*   input_seq  = (const int*)d_in[0];
    const int*   input_len  = (const int*)d_in[1];
    const int*   target_seq = (const int*)d_in[2];
    const float* enc_embed  = (const float*)d_in[3];
    const float* enc_Wih    = (const float*)d_in[4];
    const float* enc_Whh    = (const float*)d_in[5];
    const float* enc_bih    = (const float*)d_in[6];
    const float* enc_bhh    = (const float*)d_in[7];
    const float* att_W      = (const float*)d_in[8];
    const float* att_b      = (const float*)d_in[9];
    const float* dec_embed  = (const float*)d_in[10];
    const float* dec_Wih    = (const float*)d_in[11];
    const float* dec_Whh    = (const float*)d_in[12];
    const float* dec_bih    = (const float*)d_in[13];
    const float* dec_bhh    = (const float*)d_in[14];
    const float* mlp_W      = (const float*)d_in[15];
    const float* mlp_b      = (const float*)d_in[16];
    const float* out_W      = (const float*)d_in[17];
    const float* out_b      = (const float*)d_in[18];
    float* out = (float*)d_out;

    const int B  = in_sizes[1];
    const int S  = in_sizes[0] / B;
    const int T  = in_sizes[2] / B;
    const int H  = in_sizes[6] / 3;
    const int V  = in_sizes[18];
    const int H3 = 3 * H;
    const int H2 = 2 * H;

    // -------- workspace carve (everything 256B aligned) --------
    char* p = (char*)d_ws;
    auto alloc = [&](size_t bytes) -> void* {
        void* r = (void*)p;
        p += (bytes + 255) & ~(size_t)255;
        return r;
    };
    _Float16* wEncIh = (_Float16*)alloc((size_t)H3 * H * 2);
    _Float16* wEncHh = (_Float16*)alloc((size_t)H3 * H * 2);
    _Float16* wDecIh = (_Float16*)alloc((size_t)H3 * H * 2);
    _Float16* wDecHh = (_Float16*)alloc((size_t)H3 * H * 2);
    _Float16* wAtt   = (_Float16*)alloc((size_t)H * H * 2);
    _Float16* wMlp   = (_Float16*)alloc((size_t)H * H2 * 2);
    _Float16* wOut   = (_Float16*)alloc((size_t)V * H * 2);
    _Float16* encOut = (_Float16*)alloc((size_t)S * B * H * 2);   // masked encoder states (f16)
    float*    h      = (float*)alloc((size_t)B * H * 4);
    _Float16* h16    = (_Float16*)alloc((size_t)B * H * 2);
    float*    gi     = (float*)alloc((size_t)B * H3 * 4);
    float*    gh     = (float*)alloc((size_t)B * H3 * 4);
    _Float16* x16    = (_Float16*)alloc((size_t)B * H * 2);       // encoder input
    _Float16* ah16   = (_Float16*)alloc((size_t)B * H * 2);       // relu(h attW^T + b)
    float*    e      = (float*)alloc((size_t)B * S * 4);
    float*    attw   = (float*)alloc((size_t)B * S * 4);
    _Float16* xcat   = (_Float16*)alloc((size_t)B * H2 * 2);      // [emb | ctx]
    _Float16* xdec   = (_Float16*)alloc((size_t)B * H * 2);       // tanh(mlp)
    float*    logits = (float*)alloc((size_t)B * V * 4);
    int*      decTok = (int*)alloc((size_t)T * B * 4);
    float*    lossPerB    = (float*)alloc((size_t)B * 4);
    int*      correctPerB = (int*)alloc((size_t)B * 4);
    int*      wrong       = (int*)alloc((size_t)B * 4);

    auto cdiv = [](int a, int b) { return (a + b - 1) / b; };
    auto conv = [&](const float* src, _Float16* dst, int n) {
        cvt_f32_f16<<<cdiv(n, 256), 256, 0, stream>>>(src, dst, n);
    };
    auto gemm = [&](const _Float16* A, int lda, const _Float16* Wm, int ldw,
                    const float* bias, float* Cf, _Float16* Ch, int ldc,
                    int M, int N, int K, int act) {
        dim3 grid(M / 64, N / 128);
        wmma_gemm_f16<<<grid, 256, 0, stream>>>(A, lda, Wm, ldw, bias, Cf, Ch, ldc, K, act);
    };

    // -------- prologue --------
    conv(enc_Wih, wEncIh, H3 * H);
    conv(enc_Whh, wEncHh, H3 * H);
    conv(dec_Wih, wDecIh, H3 * H);
    conv(dec_Whh, wDecHh, H3 * H);
    conv(att_W,   wAtt,   H * H);
    conv(mlp_W,   wMlp,   H * H2);
    conv(out_W,   wOut,   V * H);
    init_state<<<cdiv(B * H, 256), 256, 0, stream>>>(h, h16, lossPerB, correctPerB, wrong, B * H, B);
    prep_dec_tokens<<<cdiv(T * B, 256), 256, 0, stream>>>(target_seq, decTok, T, B);

    // -------- encoder scan --------
    for (int t = 0; t < S; ++t) {
        gather_embed_f16<<<cdiv(B * H, 256), 256, 0, stream>>>(x16, H, 0, enc_embed,
                                                               input_seq + (size_t)t * B, H, B);
        gemm(x16, H, wEncIh, H, enc_bih, gi, nullptr, H3, B, H3, H, 0);
        gemm(h16, H, wEncHh, H, enc_bhh, gh, nullptr, H3, B, H3, H, 0);
        gru_update<<<cdiv(B * H, 256), 256, 0, stream>>>(gi, gh, h, h16,
                                                         encOut + (size_t)t * B * H,
                                                         input_len, t, B, H);
    }

    // -------- decoder scan --------
    for (int t = 0; t < T; ++t) {
        gemm(h16, H, wAtt, H, att_b, nullptr, ah16, H, B, H, H, 1);                 // relu
        att_scores<<<cdiv(B * S, 8), 256, 0, stream>>>(encOut, ah16, e, S, B, H);
        softmax_rows<<<B, 256, 0, stream>>>(e, attw, S);
        att_context<<<B, 256, 0, stream>>>(encOut, attw, xcat, S, B, H);
        gather_embed_f16<<<cdiv(B * H, 256), 256, 0, stream>>>(xcat, H2, 0, dec_embed,
                                                               decTok + (size_t)t * B, H, B);
        gemm(xcat, H2, wMlp, H2, mlp_b, nullptr, xdec, H, B, H, H2, 2);             // tanh
        gemm(xdec, H, wDecIh, H, dec_bih, gi, nullptr, H3, B, H3, H, 0);
        gemm(h16, H, wDecHh, H, dec_bhh, gh, nullptr, H3, B, H3, H, 0);
        gru_update<<<cdiv(B * H, 256), 256, 0, stream>>>(gi, gh, h, h16, nullptr, nullptr, t, B, H);
        gemm(h16, H, wOut, H, out_b, logits, nullptr, V, B, V, H, 0);
        out_softmax_stats<<<B, 256, 0, stream>>>(logits, target_seq + (size_t)t * B,
                                                 out + (size_t)t * B, lossPerB, correctPerB,
                                                 wrong, V);
    }

    finalize_kernel<<<1, 256, 0, stream>>>(lossPerB, correctPerB, wrong, out, T * B, B);
}
